// DNPU_Layer_39805756899544
// MI455X (gfx1250) — matrix operand order repaired and checked by reading, not compiled, not run on hardware
//
#include <hip/hip_runtime.h>

// DNPU mixture-of-tiny-MLPs on gfx1250 (MI455X).
// Per-node transposed GEMMs on v_wmma_f32_16x16x32_f16:
//   D'[j][b] = sum_k W[k][j] * h[b][k]   (A = W^T from LDS, B = h^T in regs)
// Activations stay register-resident across all 4 layers; D'->B' relayout via
// one shfl_xor(16) per register + packed-word cndmask selects; ReLU done once
// in f16 (v_pk_max) after packing.
// ALL biases are folded into the weights through a constant-1 channel in the
// H padding (j=90): C operands are inline zero -> no bias registers.

#define B_TOT 16384
#define NN    128
#define II    3
#define CC    4
#define EE    (II + CC)   // 7
#define HH    90
#define HP    96          // H padded to 6x16 WMMA tiles; j=90 is the "ones" channel
#define LLAY  3
#define KSTR  104         // padded halves per hidden-weight row: 208B -> conflict-free b128
#define WIN_K 16          // layer-0 row: K 0..6 = electrodes, K 7 = bias, 8..15 zero
#define WAVES 8
#define GRID_Y 16
#define TPW   (B_TOT / 16 / (GRID_Y * WAVES))   // 8 tiles of 16 rows per wave

typedef _Float16 v16h __attribute__((ext_vector_type(16)));
typedef _Float16 v8h  __attribute__((ext_vector_type(8)));
typedef _Float16 v2h  __attribute__((ext_vector_type(2)));
typedef float    v8f  __attribute__((ext_vector_type(8)));

__device__ __forceinline__ float relu1(float v) {
    return __builtin_amdgcn_fmed3f(v, 0.f, __builtin_inff());
}

__global__ __launch_bounds__(WAVES * 32)
void dnpu_wmma_kernel(const float* __restrict__ x,
                      const float* __restrict__ controls,
                      const float* __restrict__ W_in,
                      const float* __restrict__ b_in,
                      const float* __restrict__ W_hid,
                      const float* __restrict__ b_hid,
                      const float* __restrict__ W_out,
                      const float* __restrict__ b_out,
                      float* __restrict__ out)
{
    // 59904 + 3072 + 384 = 63360 bytes of static LDS
    __shared__ _Float16 sWhid[LLAY][HP][KSTR];  // [layer][j][k]; k=90 = bias column
    __shared__ _Float16 sWin[HP][WIN_K];        // [j][k]; k=7 = bias column
    __shared__ float    sWout[HP];              // j=90 holds b_out

    const int n   = blockIdx.x;
    const int tid = threadIdx.x;

    // ---- cooperative fp32 -> fp16 weight load + bias folding ----
    for (int idx = tid; idx < HP * WIN_K; idx += WAVES * 32) {
        int j = idx >> 4, k = idx & 15;
        float w = 0.f;
        if (k < EE)       w = (j < HH) ? W_in[(n * EE + k) * HH + j] : 0.f;
        else if (k == EE) w = (j < HH) ? b_in[n * HH + j] : (j == HH ? 1.f : 0.f);
        sWin[j][k] = (_Float16)w;
    }
    for (int idx = tid; idx < LLAY * HP * HP; idx += WAVES * 32) {
        int l = idx / (HP * HP);
        int r = idx - l * (HP * HP);
        int k = r / HP;
        int j = r - k * HP;                      // j minor -> coalesced global read
        float w = 0.f;
        if (k < HH && j < HH)
            w = W_hid[((size_t)(l * NN + n) * HH + k) * HH + j];
        else if (k == HH)                        // bias column fed by ones-channel
            w = (j < HH) ? b_hid[(l * NN + n) * HH + j] : (j == HH ? 1.f : 0.f);
        sWhid[l][j][k] = (_Float16)w;
    }
    for (int idx = tid; idx < HP; idx += WAVES * 32) {
        float w = 0.f;
        if (idx < HH)       w = W_out[n * HH + idx];
        else if (idx == HH) w = b_out[n];        // final bias via ones-channel
        sWout[idx] = w;
    }
    __syncthreads();

    const int wave = tid >> 5, lane = tid & 31;
    const int lane15 = lane & 15, hl = lane >> 4;
    const float c0 = controls[n * CC + 0], c1 = controls[n * CC + 1],
                c2 = controls[n * CC + 2], c3 = controls[n * CC + 3];
    const v8h  z8 = {};
    const v8f  zc = {};
    const v16h z16 = {};
    const _Float16 hz = (_Float16)0.f;
    const _Float16 ho = (_Float16)1.f;

    for (int t = 0; t < TPW; ++t) {
        const int tile = (blockIdx.y * WAVES + wave) * TPW + t;
        const int row0 = tile * 16;

        // ---- B0 = [e | 1]^T (K=8 real, zero padded to 32), B-format f16 ----
        const float* xr = x + (size_t)(row0 + lane15) * (NN * II) + n * II;
        float x0 = xr[0], x1 = xr[1], x2 = xr[2];
        v16h b0 = {};
        b0[0] = hl ? hz : (_Float16)x0;
        b0[1] = hl ? hz : (_Float16)x1;
        b0[2] = hl ? hz : (_Float16)x2;
        b0[3] = hl ? hz : (_Float16)c0;
        b0[4] = hl ? hz : (_Float16)c1;
        b0[5] = hl ? hz : (_Float16)c2;
        b0[6] = hl ? hz : (_Float16)c3;
        b0[7] = hl ? hz : ho;                    // ones-channel -> bias row

        // prefetch next tile's x rows (clamped), keep near caches
        {
            int pr = row0 + 16 + lane15;
            pr = pr < B_TOT ? pr : B_TOT - 1;
            __builtin_prefetch(x + (size_t)pr * (NN * II) + n * II, 0, 3);
        }

        // ---- layer 0: acc[jt] = W_in'^T x [e|1]^T   (C = inline 0) ----
        v8f acc[6];
        #pragma unroll
        for (int jt = 0; jt < 6; ++jt) {
            v8h lo = *(const v8h*)&sWin[16 * jt + lane15][8 * hl];
            v16h a = __builtin_shufflevector(lo, z8,
                      0, 1, 2, 3, 4, 5, 6, 7, 8, 9, 10, 11, 12, 13, 14, 15);
            acc[jt] = __builtin_amdgcn_wmma_f32_16x16x32_f16(
                          false, a, false, b0, (short)0, zc, false, false);
        }

        // ---- hidden layers: register transpose D'->B' + relu, 3-step K GEMM ----
        #pragma unroll
        for (int l = 0; l < LLAY; ++l) {
            v16h bk[3];
            #pragma unroll
            for (int ks = 0; ks < 3; ++ks) {
                float own[8], q[8], X[8];
                #pragma unroll
                for (int w = 0; w < 8; ++w) {
                    float u  = acc[2 * ks + 0][w];            // j-tile 2ks
                    float tt = acc[2 * ks + 1][w];            // j-tile 2ks+1
                    own[w] = hl ? tt : u;                     // own-half source
                    q[w]   = hl ? u  : tt;                    // value the partner needs
                }
                #pragma unroll
                for (int w = 0; w < 8; ++w) X[w] = __shfl_xor(q[w], 16, 32);
                // pack both streams to f16 pairs, then one 32-bit select per word
                unsigned po[4], px[4];
                #pragma unroll
                for (int i = 0; i < 4; ++i) {
                    v2h a2; a2[0] = (_Float16)own[2 * i]; a2[1] = (_Float16)own[2 * i + 1];
                    v2h b2; b2[0] = (_Float16)X[2 * i];   b2[1] = (_Float16)X[2 * i + 1];
                    po[i] = __builtin_bit_cast(unsigned, a2);
                    px[i] = __builtin_bit_cast(unsigned, b2);
                }
                union { unsigned u[8]; v16h v; } bb;
                #pragma unroll
                for (int i = 0; i < 4; ++i) {
                    // words 0..3 <-> K elems 0..7 (source half 0), 4..7 <-> 8..15 (half 1)
                    bb.u[i]     = hl ? px[i] : po[i];
                    bb.u[i + 4] = hl ? po[i] : px[i];
                }
                bk[ks] = __builtin_elementwise_max(bb.v, z16);   // relu, packed f16
            }
            #pragma unroll
            for (int jt = 0; jt < 6; ++jt) {
                // all three k-step A tiles in distinct registers -> no WAR stalls
                const _Float16* ap = &sWhid[l][16 * jt + lane15][16 * hl];
                v8h a0lo = *(const v8h*)(ap +  0), a0hi = *(const v8h*)(ap +  8);
                v8h a1lo = *(const v8h*)(ap + 32), a1hi = *(const v8h*)(ap + 40);
                v8h a2lo = *(const v8h*)(ap + 64), a2hi = *(const v8h*)(ap + 72);
                v16h a0 = __builtin_shufflevector(a0lo, a0hi,
                           0, 1, 2, 3, 4, 5, 6, 7, 8, 9, 10, 11, 12, 13, 14, 15);
                v16h a1 = __builtin_shufflevector(a1lo, a1hi,
                           0, 1, 2, 3, 4, 5, 6, 7, 8, 9, 10, 11, 12, 13, 14, 15);
                v16h a2 = __builtin_shufflevector(a2lo, a2hi,
                           0, 1, 2, 3, 4, 5, 6, 7, 8, 9, 10, 11, 12, 13, 14, 15);
                v8f c = __builtin_amdgcn_wmma_f32_16x16x32_f16(
                            false, a0, false, bk[0], (short)0, zc, false, false);
                c = __builtin_amdgcn_wmma_f32_16x16x32_f16(
                        false, a1, false, bk[1], (short)0, c, false, false);
                c = __builtin_amdgcn_wmma_f32_16x16x32_f16(
                        false, a2, false, bk[2], (short)0, c, false, false);
                acc[jt] = c;   // bias arrived via k=90 column x ones-channel
            }
        }

        // ---- output: out[b] = relu(h) . W_out'  (includes b_out via j=90) ----
        float p = 0.f;
        #pragma unroll
        for (int jt = 0; jt < 6; ++jt) {
            #pragma unroll
            for (int v = 0; v < 8; ++v)
                p += relu1(acc[jt][v]) * sWout[16 * jt + v + 8 * hl];
        }
        p += __shfl_xor(p, 16, 32);   // combine j-halves; every lane has the full sum
        if (hl == 0)
            out[(size_t)(row0 + lane15) * NN + n] = p;
    }
}

extern "C" void kernel_launch(void* const* d_in, const int* in_sizes, int n_in,
                              void* d_out, int out_size, void* d_ws, size_t ws_size,
                              hipStream_t stream)
{
    (void)in_sizes; (void)n_in; (void)out_size; (void)d_ws; (void)ws_size;
    const float* x        = (const float*)d_in[0];
    const float* controls = (const float*)d_in[1];
    const float* W_in     = (const float*)d_in[2];
    const float* b_in     = (const float*)d_in[3];
    const float* W_hid    = (const float*)d_in[4];
    const float* b_hid    = (const float*)d_in[5];
    const float* W_out    = (const float*)d_in[6];
    const float* b_out    = (const float*)d_in[7];

    dim3 grid(NN, GRID_Y);
    dim3 block(WAVES * 32);
    dnpu_wmma_kernel<<<grid, block, 0, stream>>>(
        x, controls, W_in, b_in, W_hid, b_hid, W_out, b_out, (float*)d_out);
}
